// AttentionBase_51376398795118
// MI455X (gfx1250) — matrix operand order repaired and checked by reading, not compiled, not run on hardware
//
#include <hip/hip_runtime.h>
#include <hip/hip_bf16.h>
#include <math.h>
#include <stdint.h>

// ---------------- problem constants (from reference) ----------------
constexpr int kB  = 8;
constexpr int kC  = 128;     // dim
constexpr int kN  = 16384;   // tokens
constexpr int kH  = 8;       // heads
constexpr int kC3 = 384;     // 3*dim

// Probe results (round 2): HAS wmma_f32_16x16x4_f32, tensor_load_to_lds,
// s_wait_tensorcnt; NO ds/global tr16 builtins. Both confirmed through codegen
// in round 3.
#if defined(__HIP_DEVICE_COMPILE__) && \
    __has_builtin(__builtin_amdgcn_tensor_load_to_lds) && \
    __has_builtin(__builtin_amdgcn_s_wait_tensorcnt)
#define HAS_TDM 1
#else
#define HAS_TDM 0
#endif

#if defined(__HIP_DEVICE_COMPILE__) && \
    __has_builtin(__builtin_amdgcn_wmma_f32_16x16x4_f32)
#define HAS_WMMA_F32 1
#else
#define HAS_WMMA_F32 0
#endif

// ---------------- WMMA vector types ----------------
typedef __attribute__((ext_vector_type(16))) __bf16 v16bf;
typedef __attribute__((ext_vector_type(8)))  float  v8f;
typedef __attribute__((ext_vector_type(2)))  float  v2f;
typedef __attribute__((ext_vector_type(4)))  unsigned int v4u;
typedef __attribute__((ext_vector_type(8)))  int    v8i;
typedef __attribute__((ext_vector_type(4)))  int    v4i;

union BFV {
  v16bf v;
  unsigned short u[16];
  unsigned int   d[8];
};

__device__ __forceinline__ unsigned short f2bf(float f) {
  unsigned u = __float_as_uint(f);
  u += 0x7FFFu + ((u >> 16) & 1u);          // round-to-nearest-even
  return (unsigned short)(u >> 16);
}
__device__ __forceinline__ float bf2f(unsigned int s) {
  return __uint_as_float(s << 16);
}

// A-matrix (16x32 bf16) K-pair base for VGPR j, half-wave hl (ISA 7.12.2)
__device__ __forceinline__ int a_kb(int j, int hl) {
  return (j < 4 ? 2 * j : 16 + 2 * (j - 4)) + 8 * hl;
}

#define PANEL_ROWS 32

// ---------------- generic LDS panel staging (32 rows x W cols) -----------
// scalar fallback (edges / non-TDM builds)
template <int W>
__device__ __forceinline__ void stage_scalar(unsigned short* panel,
                                             const unsigned short* grow0,
                                             int col0) {
  for (int c = threadIdx.x; c < PANEL_ROWS * (W / 8); c += 256) {
    const int row = c / (W / 8);
    const int i8  = (c % (W / 8)) * 8;
    const int gcol = col0 + i8;
#pragma unroll
    for (int e = 0; e < 8; ++e) {
      const int col = gcol + e;
      panel[row * W + i8 + e] =
          (col >= 0 && col < kN) ? grow0[(size_t)row * kN + col]
                                 : (unsigned short)0;
    }
  }
}

#if HAS_TDM
// Whole-panel 2D TDM load; right edge zero-fills via tensor_dim0.
template <int W>
__device__ __forceinline__ void stage_tdm(unsigned short* panel,
                                          const unsigned short* grow0,
                                          int col0) {
  if (threadIdx.x < 32) {                     // one wave issues (EXEC ignored)
    const unsigned lds = (unsigned)(uintptr_t)panel;          // LDS byte addr
    const unsigned long long ga =
        (unsigned long long)(uintptr_t)(grow0 + col0);        // global bytes
    // ---- D# group 0 (ISA 8.3): count=1, lds_addr, global_addr, type=2
    v4u g0;
    g0.x = 1u;
    g0.y = lds;
    g0.z = (unsigned)ga;
    g0.w = ((unsigned)((ga >> 32) & 0x01ffffffull)) | (2u << 30);
    // ---- D# group 1 (ISA 8.4): data_size=2B, dims/strides, tile W x 32
    const unsigned td0 = (unsigned)(kN - col0);   // tensor_dim0 from tile start
    v8i g1;
    g1[0] = (int)(1u << 16);                       // data_size code 1 (2 bytes)
    g1[1] = (int)((td0 & 0xffffu) << 16);          // tensor_dim0[15:0]
    g1[2] = (int)((td0 >> 16) & 0xffffu) | (int)(32u << 16); // td0 hi | td1 lo
    g1[3] = (int)((unsigned)W << 16);              // td1 hi=0 | tile_dim0=W
    g1[4] = 32;                                    // tile_dim1=32, tile_dim2=0
    g1[5] = 16384;                                 // tensor_dim0_stride lo32
    g1[6] = 0;
    g1[7] = 0;
    v4i z4 = {0, 0, 0, 0};
#if __clang_major__ >= 23
    v8i z8 = {0, 0, 0, 0, 0, 0, 0, 0};
    __builtin_amdgcn_tensor_load_to_lds(g0, g1, z4, z4, z8, 0);
#else
    __builtin_amdgcn_tensor_load_to_lds(g0, g1, z4, z4, 0);
#endif
    __builtin_amdgcn_s_wait_tensorcnt(0);
  }
}
#endif // HAS_TDM

// ---------------- K0a: convert / transpose weights to bf16 ----------------
__global__ void k_convert_weights(const float* __restrict__ w1,
                                  const float* __restrict__ w2,
                                  const float* __restrict__ wp,
                                  unsigned short* __restrict__ w1b,
                                  unsigned short* __restrict__ w2b,
                                  unsigned short* __restrict__ wpb) {
  int i = blockIdx.x * blockDim.x + threadIdx.x;
  if (i < kC3 * kC) w1b[i] = f2bf(w1[i]);
  if (i < kC * kC)  wpb[i] = f2bf(wp[i]);
  if (i < 3 * kC3 * kC3) {
    // output layout [t][co][ci]; input w_qkv2 is [co][ci][t]
    int t  = i / (kC3 * kC3);
    int r  = i % (kC3 * kC3);
    int co = r / kC3, ci = r % kC3;
    w2b[i] = f2bf(w2[(co * kC3 + ci) * 3 + t]);
  }
}

// ---------------- K0b: one-time x f32 -> bf16 (4-wide) --------------------
__global__ void k_convert_x(const float* __restrict__ x,
                            unsigned short* __restrict__ xb) {
  const int i = blockIdx.x * blockDim.x + threadIdx.x;   // group of 4 elems
  const int total4 = kB * kC * kN / 4;
  if (i < total4) {
    const float4 f = ((const float4*)x)[i];
    const unsigned long long pack =
        (unsigned long long)f2bf(f.x)         |
        ((unsigned long long)f2bf(f.y) << 16) |
        ((unsigned long long)f2bf(f.z) << 32) |
        ((unsigned long long)f2bf(f.w) << 48);
    ((unsigned long long*)xb)[i] = pack;
  }
}

// ---------------- K1: qkv1 (1x1 conv) GEMM, TDM-staged, bf16 WMMA --------
// y1[b,co,n] = sum_ci w1[co,ci] * x[b,ci,n]   M=384 K=128 N=16384
// 256 thr (8 waves); block tile 16 x 128; panel 32 x 128 (no halo).
__global__ void k_qkv1(const unsigned short* __restrict__ xb,
                       const unsigned short* __restrict__ w1b,
                       unsigned short* __restrict__ y1b) {
  __shared__ __attribute__((aligned(16))) unsigned short panel[PANEL_ROWS * 128];
  const int tid = threadIdx.x;
  const int w = tid >> 5, l = tid & 31, hl = l >> 4, lm = l & 15;
  const int m0 = blockIdx.x * 16, nblk = blockIdx.y * 128, b = blockIdx.z;
  const int nw = nblk + 16 * w;
  v8f acc = {};
  for (int k0 = 0; k0 < kC; k0 += 32) {
    const unsigned short* grow0 = xb + (size_t)(b * kC + k0) * kN;
#if HAS_TDM
    stage_tdm<128>(panel, grow0, nblk);
#else
    stage_scalar<128>(panel, grow0, nblk);
#endif
    __syncthreads();
    BFV A, Bm;
#pragma unroll
    for (int j = 0; j < 8; ++j)
      A.d[j] = *(const unsigned int*)(w1b + (m0 + lm) * kC + k0 + a_kb(j, hl));
#pragma unroll
    for (int e = 0; e < 16; ++e)
      Bm.u[e] = panel[(8 * hl + e + (e & 8)) * 128 + 16 * w + lm];
    acc = __builtin_amdgcn_wmma_f32_16x16x32_bf16(false, A.v, false, Bm.v,
                                                  (short)0, acc, false, false);
    __syncthreads();
  }
#pragma unroll
  for (int j = 0; j < 8; ++j)
    y1b[(b * kC3 + m0 + j + 8 * hl) * kN + nw + lm] = f2bf(acc[j]);
}

// ================= K2: qkv2 (3-tap conv) GEMM, TDM-staged B ===============
// LDS panel: 32 ci-rows x 144 cols (cols = n_blk-8 .. n_blk+135), stride 144.
// All 3 taps read the same staged panel at col offsets +7/+8/+9.
__global__ void k_qkv2(const unsigned short* __restrict__ y1b,
                       const unsigned short* __restrict__ w2b,
                       unsigned short* __restrict__ qkvb,
                       float* __restrict__ norms) {
  __shared__ __attribute__((aligned(16))) unsigned short panel[PANEL_ROWS * 144];
  const int tid = threadIdx.x;
  const int w = tid >> 5, l = tid & 31, hl = l >> 4, lm = l & 15;
  const int m0   = blockIdx.x * 16;
  const int nblk = blockIdx.y * 128;
  const int b    = blockIdx.z;
  const int nw   = nblk + 16 * w;              // this wave's 16-col subtile
  const int col0 = nblk - 8;                   // panel col 0 in global coords
  v8f acc = {};

  for (int k0 = 0; k0 < kC3; k0 += 32) {
    const unsigned short* grow0 = y1b + (size_t)(b * kC3 + k0) * kN;
#if HAS_TDM
    if (col0 >= 0) stage_tdm<144>(panel, grow0, col0);
    else           stage_scalar<144>(panel, grow0, col0);
#else
    stage_scalar<144>(panel, grow0, col0);
#endif
    __syncthreads();

#pragma unroll
    for (int t = 0; t < 3; ++t) {
      const unsigned short* wt = w2b + t * kC3 * kC3;
      BFV A, Bm;
#pragma unroll
      for (int j = 0; j < 8; ++j)
        A.d[j] = *(const unsigned int*)(wt + (m0 + lm) * kC3 + k0 + a_kb(j, hl));
#pragma unroll
      for (int e = 0; e < 16; ++e)
        Bm.u[e] = panel[(8 * hl + e + (e & 8)) * 144 + 16 * w + lm + 7 + t];
      acc = __builtin_amdgcn_wmma_f32_16x16x32_bf16(false, A.v, false, Bm.v,
                                                    (short)0, acc, false, false);
    }
    __syncthreads();
  }

#pragma unroll
  for (int j = 0; j < 8; ++j) {
    const int row = m0 + j + 8 * hl;
    const float val = acc[j];
    qkvb[(b * kC3 + row) * kN + nw + lm] = f2bf(val);
    float s = val * val;                       // row sum-of-squares
#pragma unroll
    for (int m = 1; m < 16; m <<= 1) s += __shfl_xor(s, m, 32);
    if (lm == 0 && row < 2 * kC)
      atomicAdd(&norms[b * kC3 + row], s);
  }
}

// ---------------- K3: finalize norms -> inverse norms ----------------
__global__ void k_norm_fin(float* __restrict__ norms) {
  int i = blockIdx.x * blockDim.x + threadIdx.x;
  if (i < kB * kC3) norms[i] = 1.0f / fmaxf(sqrtf(norms[i]), 1e-12f);
}

// ---------------- K4: partial channel-attention 16x16, contract over N ----
__global__ void k_attn(const unsigned short* __restrict__ qkvb,
                       const float* __restrict__ norms,
                       float* __restrict__ attn) {
  const int l = threadIdx.x, hl = l >> 4, lm = l & 15;
  const int bh = blockIdx.y, b = bh >> 3, h = bh & 7;
  const int c0 = blockIdx.x * 512;
  const unsigned short* q = qkvb + (size_t)(b * kC3 + h * 16) * kN;
  const unsigned short* k = qkvb + (size_t)(b * kC3 + kC + h * 16) * kN;
  v8f acc = {};
#if HAS_WMMA_F32
  for (int kk = c0; kk < c0 + 512; kk += 4) {
    unsigned aq = *(const unsigned int*)(q + lm * kN + kk + 2 * hl);
    unsigned bk = *(const unsigned int*)(k + lm * kN + kk + 2 * hl);
    v2f a, bb;
    a.x  = bf2f(aq & 0xffffu); a.y  = bf2f(aq >> 16);
    bb.x = bf2f(bk & 0xffffu); bb.y = bf2f(bk >> 16);
    acc = __builtin_amdgcn_wmma_f32_16x16x4_f32(false, a, false, bb,
                                                (short)0, acc, false, false);
  }
#else
  for (int kk = c0; kk < c0 + 512; kk += 32) {
    BFV A, Bm;
#pragma unroll
    for (int j = 0; j < 8; ++j)
      A.d[j] = *(const unsigned int*)(q + lm * kN + kk + a_kb(j, hl));
#pragma unroll
    for (int e = 0; e < 16; ++e)
      Bm.u[e] = k[lm * kN + kk + 16 * hl + e];
    acc = __builtin_amdgcn_wmma_f32_16x16x32_bf16(false, A.v, false, Bm.v,
                                                  (short)0, acc, false, false);
  }
#endif
#pragma unroll
  for (int j = 0; j < 8; ++j) {
    const int row = j + 8 * hl, col = lm;
    const float v = acc[j] * norms[b * kC3 + h * 16 + row]
                           * norms[b * kC3 + kC + h * 16 + col];
    atomicAdd(&attn[(bh * 16 + row) * 16 + col], v);
  }
}

// ---------------- K5: scale + softmax over last dim ----------------
__global__ void k_softmax(float* __restrict__ attn, const float* __restrict__ scale) {
  const int bh = blockIdx.x, h = bh & 7;
  const int r = threadIdx.x;
  if (r >= 16) return;
  const float sc = scale[h];
  float* row = attn + (bh * 16 + r) * 16;
  float vals[16], m = -3.4e38f;
  for (int d = 0; d < 16; ++d) { vals[d] = row[d] * sc; m = fmaxf(m, vals[d]); }
  float s = 0.f;
  for (int d = 0; d < 16; ++d) { vals[d] = __expf(vals[d] - m); s += vals[d]; }
  const float inv = 1.0f / s;
  for (int d = 0; d < 16; ++d) row[d] = vals[d] * inv;
}

// ---------------- K6: out = attn @ v  (K=16) ----------------
__global__ void k_out(const float* __restrict__ attn,
                      const unsigned short* __restrict__ qkvb,
                      unsigned short* __restrict__ outb) {
  const int l = threadIdx.x, hl = l >> 4, lm = l & 15;
  const int n0 = blockIdx.x * 16, h = blockIdx.y, b = blockIdx.z;
  const int bh = b * kH + h;
  const unsigned short* v = qkvb + (size_t)(b * kC3 + 2 * kC + h * 16) * kN;
  v8f acc = {};
#if HAS_WMMA_F32
#pragma unroll
  for (int kk = 0; kk < 16; kk += 4) {
    const int d0 = kk + 2 * hl;
    v2f a, bb;
    a.x  = attn[(bh * 16 + lm) * 16 + d0];
    a.y  = attn[(bh * 16 + lm) * 16 + d0 + 1];
    bb.x = bf2f((unsigned)v[(size_t)d0 * kN + n0 + lm]);
    bb.y = bf2f((unsigned)v[(size_t)(d0 + 1) * kN + n0 + lm]);
    acc = __builtin_amdgcn_wmma_f32_16x16x4_f32(false, a, false, bb,
                                                (short)0, acc, false, false);
  }
#else
  {
    BFV A, Bm;                                  // K=32 with upper half zero
#pragma unroll
    for (int j = 0; j < 8; ++j) {
      const int d0 = a_kb(j, hl);
      if (d0 < 16) {
        unsigned lo = (unsigned)f2bf(attn[(bh * 16 + lm) * 16 + d0]);
        unsigned hi = (unsigned)f2bf(attn[(bh * 16 + lm) * 16 + d0 + 1]);
        A.d[j] = lo | (hi << 16);
      } else A.d[j] = 0;
    }
#pragma unroll
    for (int e = 0; e < 16; ++e) {
      const int d0 = 16 * hl + e;
      Bm.u[e] = (d0 < 16) ? v[(size_t)d0 * kN + n0 + lm] : (unsigned short)0;
    }
    acc = __builtin_amdgcn_wmma_f32_16x16x32_bf16(false, A.v, false, Bm.v,
                                                  (short)0, acc, false, false);
  }
#endif
#pragma unroll
  for (int j = 0; j < 8; ++j)
    outb[(b * kC + h * 16 + j + 8 * hl) * kN + n0 + lm] = f2bf(acc[j]);
}

// ---------------- K7: final 1x1 proj GEMM, TDM-staged, f32 output ---------
__global__ void k_proj(const unsigned short* __restrict__ outb,
                       const unsigned short* __restrict__ wpb,
                       float* __restrict__ out) {
  __shared__ __attribute__((aligned(16))) unsigned short panel[PANEL_ROWS * 128];
  const int tid = threadIdx.x;
  const int w = tid >> 5, l = tid & 31, hl = l >> 4, lm = l & 15;
  const int m0 = blockIdx.x * 16, nblk = blockIdx.y * 128, b = blockIdx.z;
  const int nw = nblk + 16 * w;
  v8f acc = {};
  for (int k0 = 0; k0 < kC; k0 += 32) {
    const unsigned short* grow0 = outb + (size_t)(b * kC + k0) * kN;
#if HAS_TDM
    stage_tdm<128>(panel, grow0, nblk);
#else
    stage_scalar<128>(panel, grow0, nblk);
#endif
    __syncthreads();
    BFV A, Bm;
#pragma unroll
    for (int j = 0; j < 8; ++j)
      A.d[j] = *(const unsigned int*)(wpb + (m0 + lm) * kC + k0 + a_kb(j, hl));
#pragma unroll
    for (int e = 0; e < 16; ++e)
      Bm.u[e] = panel[(8 * hl + e + (e & 8)) * 128 + 16 * w + lm];
    acc = __builtin_amdgcn_wmma_f32_16x16x32_bf16(false, A.v, false, Bm.v,
                                                  (short)0, acc, false, false);
    __syncthreads();
  }
#pragma unroll
  for (int j = 0; j < 8; ++j)
    out[((size_t)b * kC + m0 + j + 8 * hl) * kN + nw + lm] = acc[j];
}

// ---------------- launch ----------------
extern "C" void kernel_launch(void* const* d_in, const int* in_sizes, int n_in,
                              void* d_out, int out_size, void* d_ws, size_t ws_size,
                              hipStream_t stream) {
  const float* x     = (const float*)d_in[0];
  const float* w1    = (const float*)d_in[1];
  const float* w2    = (const float*)d_in[2];
  const float* wp    = (const float*)d_in[3];
  const float* scale = (const float*)d_in[4];

  char* ws = (char*)d_ws;
  unsigned short* w1b  = (unsigned short*)(ws + 0);                      //  96 KB
  unsigned short* w2b  = (unsigned short*)(ws + 98304);                  // 864 KB
  unsigned short* wpb  = (unsigned short*)(ws + 983040);                 //  32 KB
  float*          norms= (float*)(ws + 1015808);                         //  12 KB
  float*          attn = (float*)(ws + 1028096);                         //  64 KB
  unsigned short* y1b  = (unsigned short*)(ws + 1093632);                //  96 MB
  unsigned short* qkvb = (unsigned short*)(ws + 1093632 + 100663296);    //  96 MB
  unsigned short* outb = (unsigned short*)(ws + 1093632 + 201326592);    //  32 MB
  unsigned short* xb   = (unsigned short*)(ws + 1093632 + 234881024);    //  32 MB

  (void)hipMemsetAsync(norms, 0, (size_t)kB * kC3 * sizeof(float), stream);
  (void)hipMemsetAsync(attn,  0, (size_t)kB * kH * 256 * sizeof(float), stream);

  k_convert_weights<<<dim3((3 * kC3 * kC3 + 255) / 256), 256, 0, stream>>>(
      w1, w2, wp, w1b, w2b, wpb);
  k_convert_x<<<dim3(kB * kC * kN / 4 / 256), 256, 0, stream>>>(x, xb);
  k_qkv1<<<dim3(kC3 / 16, kN / 128, kB), 256, 0, stream>>>(xb, w1b, y1b);
  k_qkv2<<<dim3(kC3 / 16, kN / 128, kB), 256, 0, stream>>>(y1b, w2b, qkvb, norms);
  k_norm_fin<<<dim3((kB * kC3 + 255) / 256), 256, 0, stream>>>(norms);
  k_attn<<<dim3(kN / 512, kB * kH), 32, 0, stream>>>(qkvb, norms, attn);
  k_softmax<<<dim3(kB * kH), 32, 0, stream>>>(attn, scale);
  k_out<<<dim3(kN / 16, kH, kB), 32, 0, stream>>>(attn, qkvb, outb);
  k_proj<<<dim3(kC / 16, kN / 128, kB), 256, 0, stream>>>(outb, wpb, (float*)d_out);
}
